// Biaffine_60773787238794
// MI455X (gfx1250) — matrix operand order repaired and checked by reading, not compiled, not run on hardware
//
#include <hip/hip_runtime.h>
#include <hip/hip_bf16.h>
#include <cstdint>

#define B_   32
#define CF   128
#define OUT_ 128
#define LL   8192
#define ST   136   // bf16 row stride (elements)
#define ST2  132   // packed-uint row stride (uints)

typedef __attribute__((ext_vector_type(16))) __bf16 v16bf;
typedef __attribute__((ext_vector_type(8)))  float  v8f;
typedef __attribute__((ext_vector_type(4)))  int    v4i;

#if defined(__has_builtin)
#if __has_builtin(__builtin_amdgcn_global_load_async_to_lds_b128)
#define USE_ASYNC 1
#endif
#endif

#ifdef USE_ASYNC
typedef __attribute__((address_space(1))) v4i gv4i;   // global int4
typedef __attribute__((address_space(3))) v4i lv4i;   // LDS int4
__device__ __forceinline__ gv4i* to_glb(const void* p) {
    return (gv4i*)(uintptr_t)p;
}
__device__ __forceinline__ lv4i* to_lds(void* p) {
    return (lv4i*)(unsigned)(uintptr_t)p;   // generic->LDS: low 32 bits
}
#endif

__device__ __forceinline__ void wait_async0() {
#if defined(__has_builtin)
#if __has_builtin(__builtin_amdgcn_s_wait_asynccnt)
    __builtin_amdgcn_s_wait_asynccnt(0);
    return;
#endif
#endif
    asm volatile("s_wait_asynccnt 0x0" ::: "memory");
}

// scheduling fence: keep fragment loads above, WMMA burst below
__device__ __forceinline__ void sched_split() {
#if defined(__has_builtin)
#if __has_builtin(__builtin_amdgcn_sched_barrier)
    __builtin_amdgcn_sched_barrier(0);
#endif
#endif
}

// ---- WMMA: D = A(16x32 bf16) * B(32x16 bf16) + C(16x16 f32) ----
__device__ __forceinline__ v8f wmma_bf16(v16bf a, v16bf b, v8f c) {
    return __builtin_amdgcn_wmma_f32_16x16x32_bf16(false, a, false, b, (short)0, c, false, false);
}

__device__ __forceinline__ unsigned pack2(float lo, float hi) {
    unsigned short l = __builtin_bit_cast(unsigned short, (__bf16)lo);
    unsigned short h = __builtin_bit_cast(unsigned short, (__bf16)hi);
    return ((unsigned)h << 16) | (unsigned)l;
}

// A fragment (16x32 MxK) from row-major bf16 LDS [m][k], stride ST.
__device__ __forceinline__ v16bf load_frag_A(const __bf16* lds, int row0, int k0, int lane) {
    v16bf f;
    int m  = lane & 15;
    int kb = (lane >> 4) * 16;
    const __bf16* base = lds + (row0 + m) * ST;
#pragma unroll
    for (int v = 0; v < 8; ++v) {
        int k = k0 + kb + v * 2;
        ((unsigned*)&f)[v] = *(const unsigned*)(base + k);
    }
    return f;
}

// B fragment (32x16 KxN) from K-pair-packed uint LDS [k/2][n], stride ST2.
__device__ __forceinline__ v16bf load_frag_Bp(const unsigned* ldsu, int k0, int n0, int lane) {
    v16bf f;
    int n   = n0 + (lane & 15);
    int kp0 = (k0 + (lane >> 4) * 16) >> 1;
    const unsigned* base = ldsu + (size_t)kp0 * ST2 + n;
#pragma unroll
    for (int v = 0; v < 8; ++v)
        ((unsigned*)&f)[v] = base[(size_t)v * ST2];
    return f;
}

// B fragment (32x16 KxN) from row-major bf16 LDS [k][n], stride ST (16-bit reads).
__device__ __forceinline__ v16bf load_frag_B(const __bf16* lds, int k0, int n0, int lane) {
    v16bf f;
    int n  = n0 + (lane & 15);
    int kb = k0 + (lane >> 4) * 16;
#pragma unroll
    for (int v = 0; v < 8; ++v) {
        int k = kb + 2 * v;
        unsigned lo = *(const unsigned short*)(lds + (size_t)k * ST + n);
        unsigned hi = *(const unsigned short*)(lds + (size_t)(k + 1) * ST + n);
        ((unsigned*)&f)[v] = lo | (hi << 16);
    }
    return f;
}

// B[k][n] = M[n][k] with M row-major bf16 [n][k]: contiguous K-pair -> b32 reads.
__device__ __forceinline__ v16bf load_frag_BT(const __bf16* lds, int n0, int k0, int lane) {
    v16bf f;
    int n  = n0 + (lane & 15);
    int kb = k0 + (lane >> 4) * 16;
    const __bf16* base = lds + (size_t)n * ST;
#pragma unroll
    for (int v = 0; v < 8; ++v)
        ((unsigned*)&f)[v] = *(const unsigned*)(base + kb + 2 * v);
    return f;
}

// ================= Kernel 1: projections + z + split-K scores ==============
// 8 waves tile the 128x128 output as 4 row-groups (32) x 2 col-groups (64).
__global__ __launch_bounds__(256) void k1_proj_scores(
    const float* __restrict__ x, const float* __restrict__ y,
    const float* __restrict__ Wf, const float* __restrict__ bfv,
    const float* __restrict__ Wa, const float* __restrict__ bav,
    __bf16* __restrict__ zout, float* __restrict__ sp /* [16][B][128][128] */)
{
    extern __shared__ __bf16 smem[];
    __bf16* sW0 = smem;                 // Wf bf16 (persistent)
    __bf16* sW1 = sW0 + 128 * ST;       // Wa bf16 (persistent)
    __bf16* sT0 = sW1 + 128 * ST;       // phase1: packed x-tile; phase2: xp row-major
    __bf16* sT1 = sT0 + 128 * ST;       // phase1: packed y-tile; phase2: yp row-major
    unsigned* sT0u = (unsigned*)sT0;
    unsigned* sT1u = (unsigned*)sT1;
    float*  sb  = (float*)(sT1 + 128 * ST); // 256 floats: bf | ba

    const int chunk = blockIdx.x, b = blockIdx.y;
    const int tid = threadIdx.x, lane = tid & 31, wave = tid >> 5;
    const int mrow0 = (wave & 3) * 32;      // wave's 32-row group
    const int ncol0 = (wave >> 2) * 64;     // wave's 64-col group
    const int hi = lane >> 4, nn = lane & 15;

    for (int i = tid; i < 128 * 128; i += 256) {
        int r = i >> 7, c = i & 127;
        sW0[r * ST + c] = (__bf16)Wf[i];
        sW1[r * ST + c] = (__bf16)Wa[i];
    }
    if (tid < 128) { sb[tid] = bfv[tid]; sb[128 + tid] = bav[tid]; }

    v8f accS[2][4] = {};   // persistent scores accumulators (32x64 tile)

    const float* xb = x + (size_t)b * CF * LL;
    const float* yb = y + (size_t)b * CF * LL;
    __bf16* zb = zout + (size_t)b * OUT_ * LL;

    for (int s = 0; s < 4; ++s) {
        const int l0 = (chunk * 4 + s) * 128;
        __syncthreads();
        // stage x,y as K-pair packed bf16 pairs: sTu[kp][n] = {lo: row 2kp, hi: row 2kp+1}
        for (int i = tid; i < 64 * 32; i += 256) {
            int kp = i >> 5, c4 = (i & 31) * 4;
            const float* xr0 = xb + (size_t)(2 * kp) * LL + l0 + c4;
            const float* yr0 = yb + (size_t)(2 * kp) * LL + l0 + c4;
            float4 a0 = *(const float4*)xr0;
            float4 a1 = *(const float4*)(xr0 + LL);
            float4 b0 = *(const float4*)yr0;
            float4 b1 = *(const float4*)(yr0 + LL);
            unsigned* dx = sT0u + (size_t)kp * ST2 + c4;
            dx[0] = pack2(a0.x, a1.x); dx[1] = pack2(a0.y, a1.y);
            dx[2] = pack2(a0.z, a1.z); dx[3] = pack2(a0.w, a1.w);
            unsigned* dy = sT1u + (size_t)kp * ST2 + c4;
            dy[0] = pack2(b0.x, b1.x); dy[1] = pack2(b0.y, b1.y);
            dy[2] = pack2(b0.z, b1.z); dy[3] = pack2(b0.w, b1.w);
            if (s < 3 && (c4 & 12) == 0) {       // prefetch next subtile, 64B grain
                __builtin_prefetch(xr0 + 128, 0, 1);
                __builtin_prefetch(xr0 + LL + 128, 0, 1);
                __builtin_prefetch(yr0 + 128, 0, 1);
                __builtin_prefetch(yr0 + LL + 128, 0, 1);
            }
        }
        __syncthreads();

        // ---- xp tile = Wf[mrow0:+32,:] @ x-tile[:, ncol0:+64] ----
        v8f acc[2][4] = {};
#pragma unroll
        for (int kt = 0; kt < 4; ++kt) {
            v16bf a0 = load_frag_A(sW0, mrow0,      kt * 32, lane);
            v16bf a1 = load_frag_A(sW0, mrow0 + 16, kt * 32, lane);
            v16bf bb[4];
#pragma unroll
            for (int q = 0; q < 4; ++q)
                bb[q] = load_frag_Bp(sT0u, kt * 32, ncol0 + q * 16, lane);
            sched_split();
#pragma unroll
            for (int q = 0; q < 4; ++q) {
                acc[0][q] = wmma_bf16(a0, bb[q], acc[0][q]);
                acc[1][q] = wmma_bf16(a1, bb[q], acc[1][q]);
            }
        }
        __syncthreads();                       // all waves done reading sT0
#pragma unroll
        for (int mi = 0; mi < 2; ++mi)
#pragma unroll
            for (int q = 0; q < 4; ++q)
#pragma unroll
                for (int r = 0; r < 8; ++r) {
                    int o = mrow0 + mi * 16 + r + 8 * hi;
                    float v = acc[mi][q][r] + sb[o];
                    sT0[o * ST + ncol0 + q * 16 + nn] = (__bf16)v;   // xp tile
                }

        // ---- yp tile = Wa[mrow0:+32,:] @ y-tile[:, ncol0:+64] ----
#pragma unroll
        for (int mi = 0; mi < 2; ++mi)
#pragma unroll
            for (int q = 0; q < 4; ++q) acc[mi][q] = (v8f){};
#pragma unroll
        for (int kt = 0; kt < 4; ++kt) {
            v16bf a0 = load_frag_A(sW1, mrow0,      kt * 32, lane);
            v16bf a1 = load_frag_A(sW1, mrow0 + 16, kt * 32, lane);
            v16bf bb[4];
#pragma unroll
            for (int q = 0; q < 4; ++q)
                bb[q] = load_frag_Bp(sT1u, kt * 32, ncol0 + q * 16, lane);
            sched_split();
#pragma unroll
            for (int q = 0; q < 4; ++q) {
                acc[0][q] = wmma_bf16(a0, bb[q], acc[0][q]);
                acc[1][q] = wmma_bf16(a1, bb[q], acc[1][q]);
            }
        }
        __syncthreads();                       // sT1 reads + xp writes done
#pragma unroll
        for (int mi = 0; mi < 2; ++mi)
#pragma unroll
            for (int q = 0; q < 4; ++q)
#pragma unroll
                for (int r = 0; r < 8; ++r) {
                    int o = mrow0 + mi * 16 + r + 8 * hi;
                    float v = acc[mi][q][r] + sb[128 + o];
                    sT1[o * ST + ncol0 + q * 16 + nn] = (__bf16)v;   // yp tile
                }
        __syncthreads();

        // ---- z = xp + yp -> global bf16 (coalesced in l) ----
        for (int i = tid; i < 128 * 128; i += 256) {
            int r = i >> 7, c = i & 127;
            float v = (float)sT0[r * ST + c] + (float)sT1[r * ST + c];
            zb[(size_t)r * LL + l0 + c] = (__bf16)v;
        }

        // ---- scores tile: S[mrow0:+32, ncol0:+64] += xp @ yp^T ----
#pragma unroll
        for (int kt = 0; kt < 4; ++kt) {
            v16bf a0 = load_frag_A(sT0, mrow0,      kt * 32, lane);
            v16bf a1 = load_frag_A(sT0, mrow0 + 16, kt * 32, lane);
            v16bf bb[4];
#pragma unroll
            for (int q = 0; q < 4; ++q)
                bb[q] = load_frag_BT(sT1, ncol0 + q * 16, kt * 32, lane);
            sched_split();
#pragma unroll
            for (int q = 0; q < 4; ++q) {
                accS[0][q] = wmma_bf16(a0, bb[q], accS[0][q]);
                accS[1][q] = wmma_bf16(a1, bb[q], accS[1][q]);
            }
        }
    }

    float* spb = sp + ((size_t)chunk * B_ + b) * 128 * 128;
#pragma unroll
    for (int mi = 0; mi < 2; ++mi)
#pragma unroll
        for (int q = 0; q < 4; ++q)
#pragma unroll
            for (int r = 0; r < 8; ++r)
                spb[(size_t)(mrow0 + mi * 16 + r + 8 * hi) * 128 + ncol0 + q * 16 + nn]
                    = accS[mi][q][r];
}

// ================= Kernel 2: split-K reduce + softmax ======================
__global__ __launch_bounds__(128) void k2_softmax(const float* __restrict__ sp,
                                                  __bf16* __restrict__ attn)
{
    const int row = blockIdx.x;   // b*128 + o
    const int p = threadIdx.x;
    __shared__ float red[128];
    float v = 0.f;
#pragma unroll
    for (int c = 0; c < 16; ++c)
        v += sp[((size_t)c * (B_ * 128) + row) * 128 + p];
    red[p] = v; __syncthreads();
    for (int off = 64; off > 0; off >>= 1) {
        if (p < off) red[p] = fmaxf(red[p], red[p + off]);
        __syncthreads();
    }
    float m = red[0]; __syncthreads();
    float e = __expf(v - m);
    red[p] = e; __syncthreads();
    for (int off = 64; off > 0; off >>= 1) {
        if (p < off) red[p] += red[p + off];
        __syncthreads();
    }
    float a = e / (red[0] * 90.50966799187809f);   // softmax / sqrt(L)
    attn[(size_t)row * 128 + p] = (__bf16)a;
}

// ================= Kernel 3: out = attn @ z ================================
__global__ __launch_bounds__(256) void k3_out(const __bf16* __restrict__ attn,
                                              const __bf16* __restrict__ z,
                                              float* __restrict__ out)
{
    extern __shared__ __bf16 smem[];
    __bf16* sA = smem;              // attn[b] 128x128 row-major
    __bf16* sZ = sA + 128 * ST;     // z tile 128x128 row-major
    const int lt = blockIdx.x, b = blockIdx.y;
    const int tid = threadIdx.x, lane = tid & 31, wave = tid >> 5;
    const int mrow0 = (wave & 3) * 32;
    const int ncol0 = (wave >> 2) * 64;
    const int l0 = lt * 128;
    const int hi = lane >> 4, nn = lane & 15;

    const __bf16* ab = attn + (size_t)b * 128 * 128;
    const __bf16* zb = z + (size_t)b * 128 * LL;

#ifdef USE_ASYNC
    for (int i = tid; i < 128 * 16; i += 256) {          // attn via async DMA to LDS
        int r = i >> 4, c8 = (i & 15) * 8;
        __builtin_amdgcn_global_load_async_to_lds_b128(
            to_glb(ab + (size_t)r * 128 + c8), to_lds(sA + r * ST + c8), 0, 0);
    }
    for (int i = tid; i < 128 * 16; i += 256) {          // z tile via async DMA to LDS
        int r = i >> 4, c8 = (i & 15) * 8;
        __builtin_amdgcn_global_load_async_to_lds_b128(
            to_glb(zb + (size_t)r * LL + l0 + c8), to_lds(sZ + r * ST + c8), 0, 0);
    }
    wait_async0();
#else
    for (int i = tid; i < 128 * 64; i += 256) {
        int r = i >> 6, c2 = (i & 63) * 2;
        *(unsigned*)(sA + r * ST + c2) = *(const unsigned*)(ab + (size_t)r * 128 + c2);
    }
    for (int i = tid; i < 128 * 16; i += 256) {
        int r = i >> 4, c8 = (i & 15) * 8;
        *(uint4*)(sZ + r * ST + c8) = *(const uint4*)(zb + (size_t)r * LL + l0 + c8);
    }
#endif
    __syncthreads();

    v8f acc[2][4] = {};
#pragma unroll
    for (int kt = 0; kt < 4; ++kt) {
        v16bf a0 = load_frag_A(sA, mrow0,      kt * 32, lane);
        v16bf a1 = load_frag_A(sA, mrow0 + 16, kt * 32, lane);
        v16bf bb[4];
#pragma unroll
        for (int q = 0; q < 4; ++q)
            bb[q] = load_frag_B(sZ, kt * 32, ncol0 + q * 16, lane);
        sched_split();
#pragma unroll
        for (int q = 0; q < 4; ++q) {
            acc[0][q] = wmma_bf16(a0, bb[q], acc[0][q]);
            acc[1][q] = wmma_bf16(a1, bb[q], acc[1][q]);
        }
    }

    float* ob = out + (size_t)b * 128 * LL + l0;
#pragma unroll
    for (int mi = 0; mi < 2; ++mi)
#pragma unroll
        for (int q = 0; q < 4; ++q)
#pragma unroll
            for (int r = 0; r < 8; ++r)
                ob[(size_t)(mrow0 + mi * 16 + r + 8 * hi) * LL + ncol0 + q * 16 + nn]
                    = acc[mi][q][r];
}

// ================= launch ==================================================
extern "C" void kernel_launch(void* const* d_in, const int* in_sizes, int n_in,
                              void* d_out, int out_size, void* d_ws, size_t ws_size,
                              hipStream_t stream) {
    (void)in_sizes; (void)n_in; (void)out_size; (void)ws_size;
    const float* x   = (const float*)d_in[0];
    const float* y   = (const float*)d_in[1];
    const float* Wf  = (const float*)d_in[2];
    const float* bfv = (const float*)d_in[3];
    const float* Wa  = (const float*)d_in[4];
    const float* bav = (const float*)d_in[5];
    float* out = (float*)d_out;

    char* ws = (char*)d_ws;
    __bf16* z    = (__bf16*)ws;                                        // 64 MB
    float*  sp   = (float*)(ws + (size_t)B_ * OUT_ * LL * 2);          // 32 MB
    __bf16* attn = (__bf16*)((char*)sp + (size_t)16 * B_ * 128 * 128 * 4); // 1 MB

    const size_t lds1 = (size_t)4 * 128 * ST * 2 + 256 * 4;  // ~140 KB (of 320 KB/WGP)
    const size_t lds3 = (size_t)2 * 128 * ST * 2;            // ~70 KB

    k1_proj_scores<<<dim3(16, B_), 256, lds1, stream>>>(x, y, Wf, bfv, Wa, bav, z, sp);
    k2_softmax<<<dim3(B_ * 128), 128, 0, stream>>>(sp, attn);
    k3_out<<<dim3(LL / 128, B_), 256, lds3, stream>>>(attn, z, out);
}
